// Attention_2937757630492
// MI455X (gfx1250) — matrix operand order repaired and checked by reading, not compiled, not run on hardware
//
#include <hip/hip_runtime.h>

typedef __attribute__((ext_vector_type(16))) __bf16 v16bf;
typedef __attribute__((ext_vector_type(8)))  __bf16 v8bf;
typedef __attribute__((ext_vector_type(8)))  float  v8f;

#define NTOK   197
#define NHEAD  12
#define HD     64
#define DIMC   768
#define NBATCH 64
#define NPAD   224               // token dim padded to 7*32 for the PV K-loop
#define MROWS  (NBATCH * NTOK)   // 12608
#define MTILES (MROWS / 16)      // 788 (exact)

// ---------------- WMMA wrapper ----------------
__device__ __forceinline__ v8f wmma_bf(v16bf a, v16bf b, v8f c) {
  return __builtin_amdgcn_wmma_f32_16x16x32_bf16(
      /*neg_a=*/false, a, /*neg_b=*/false, b,
      /*c_mod=*/(short)0, c, /*reuse_a=*/false, /*reuse_b=*/false);
}

__device__ __forceinline__ v8f zero8() {
  return (v8f){0.f, 0.f, 0.f, 0.f, 0.f, 0.f, 0.f, 0.f};
}

// ---------------- Fragment builders (ISA 7.12.2 layouts, wave32) ----------------
// A (16x32 bf16): lane holds row M=lane&15; element j of v16bf covers
//   K = kbase + (j>>3)*16 + (lane>>4)*8 + (j&7)  -> two contiguous 8-elem chunks.
// rowp must be dereferenceable even when !valid (caller clamps the row).
__device__ __forceinline__ v16bf a_frag_bf16(const __bf16* __restrict__ rowp,
                                             int kbase, int lane, bool valid) {
  const int h8 = (lane >> 4) << 3;
  v8bf lo = *(const v8bf*)(rowp + kbase + h8);        // 16B aligned
  v8bf hi = *(const v8bf*)(rowp + kbase + 16 + h8);
  v16bf a;
#pragma unroll
  for (int j = 0; j < 8; ++j) { a[j] = lo[j]; a[8 + j] = hi[j]; }
  if (!valid) {
#pragma unroll
    for (int j = 0; j < 16; ++j) a[j] = (__bf16)0.f;
  }
  return a;
}

// B (32x16 bf16): lane holds column N=lane&15; element j covers
//   K = kbase + (j>>3)*16 + (lane>>4)*8 + (j&7)  -> two contiguous 8-elem chunks.
// Source is "row-major by N": element(k,n) = src[n*ldk + k].
__device__ __forceinline__ v16bf b_frag_bf16(const __bf16* __restrict__ src, int ldk,
                                             int n, int kbase, int lane, bool valid) {
  const __bf16* p = src + (size_t)n * ldk + kbase + ((lane >> 4) << 3);
  v8bf c0 = *(const v8bf*)(p);       // 16B aligned
  v8bf c1 = *(const v8bf*)(p + 16);
  v16bf b;
#pragma unroll
  for (int j = 0; j < 8; ++j) { b[j] = c0[j]; b[8 + j] = c1[j]; }
  if (!valid) {
#pragma unroll
    for (int j = 0; j < 16; ++j) b[j] = (__bf16)0.f;
  }
  return b;
}

// ---------------- Kernel C: f32 -> bf16 bulk convert (8 elems/thread) ----------------
__global__ __launch_bounds__(256) void cvt_kernel(const float* __restrict__ src,
                                                  __bf16* __restrict__ dst, int n) {
  int i = (blockIdx.x * 256 + threadIdx.x) * 8;
  if (i >= n) return;
  float4 f0 = *(const float4*)(src + i);
  float4 f1 = *(const float4*)(src + i + 4);
  v8bf o;
  o[0] = (__bf16)f0.x; o[1] = (__bf16)f0.y; o[2] = (__bf16)f0.z; o[3] = (__bf16)f0.w;
  o[4] = (__bf16)f1.x; o[5] = (__bf16)f1.y; o[6] = (__bf16)f1.z; o[7] = (__bf16)f1.w;
  *(v8bf*)(dst + i) = o;
}

// ---------------- Kernel 0: zero V^T padding columns (tokens 197..223) ----------------
__global__ __launch_bounds__(256) void pad_kernel(__bf16* __restrict__ vT) {
  const int PADW = NPAD - NTOK;                       // 27
  const int total = NBATCH * NHEAD * HD * PADW;       // 1,327,104
  int idx = blockIdx.x * 256 + threadIdx.x;
  if (idx >= total) return;
  int c = idx % PADW;
  int row = idx / PADW;                               // bh*64 + d
  vT[(size_t)row * NPAD + NTOK + c] = (__bf16)0.f;
}

// ---------------- Kernel 1: QKV projection + scatter ----------------
// C[12608, 2304] = x @ qkv_w^T + bias ; scatter into q(bf16,scaled), k(natural), v^T
// Each wave: 2 m-tiles x 4 n-tiles (8 WMMAs per K-step).
__global__ __launch_bounds__(128) void qkv_kernel(
    const __bf16* __restrict__ xb, const __bf16* __restrict__ wqkv,
    const float* __restrict__ q_bias, const float* __restrict__ v_bias,
    __bf16* __restrict__ qs, __bf16* __restrict__ kn, __bf16* __restrict__ vT) {
  const int lane = threadIdx.x & 31;
  const int w    = threadIdx.x >> 5;
  const int mt0  = blockIdx.x * 2;                   // 2 m-tiles, 0..787
  const int nt0  = (blockIdx.y * 4 + w) * 4;         // 4 n-tiles per wave
  const int l15  = lane & 15;
  const int h4   = lane >> 4;

  v8f acc[2][4];
#pragma unroll
  for (int m = 0; m < 2; ++m)
#pragma unroll
    for (int t = 0; t < 4; ++t) acc[m][t] = zero8();

  const __bf16* xrow0 = xb + (size_t)(mt0 * 16 + l15) * DIMC;
  const __bf16* xrow1 = xrow0 + (size_t)16 * DIMC;
  for (int kt = 0; kt < DIMC / 32; ++kt) {
    v16bf a0 = a_frag_bf16(xrow0, kt * 32, lane, true);
    v16bf a1 = a_frag_bf16(xrow1, kt * 32, lane, true);
#pragma unroll
    for (int t = 0; t < 4; ++t) {
      int n = (nt0 + t) * 16 + l15;
      v16bf b = b_frag_bf16(wqkv, DIMC, n, kt * 32, lane, true);
      acc[0][t] = wmma_bf(a0, b, acc[0][t]);
      acc[1][t] = wmma_bf(a1, b, acc[1][t]);
    }
  }

#pragma unroll
  for (int m = 0; m < 2; ++m)
#pragma unroll
    for (int t = 0; t < 4; ++t) {
      int c     = (nt0 + t) * 16 + l15;   // 0..2303
      int which = c / DIMC;               // 0=q 1=k 2=v
      int rem   = c % DIMC;
      int hh    = rem >> 6;
      int d     = rem & 63;
      float bias = (which == 0) ? q_bias[rem] : (which == 2 ? v_bias[rem] : 0.f);
#pragma unroll
      for (int r = 0; r < 8; ++r) {
        int tok = (mt0 + m) * 16 + r + 8 * h4;   // < 12608 always
        int b_  = tok / NTOK;
        int n_  = tok % NTOK;
        float val = acc[m][t][r] + bias;
        size_t bh = (size_t)(b_ * NHEAD + hh);
        if (which == 0) {
          qs[(bh * NTOK + n_) * HD + d] = (__bf16)(val * 0.125f);  // scale = hd^-0.5
        } else if (which == 1) {
          kn[(bh * NTOK + n_) * HD + d] = (__bf16)val;             // natural [tok, 64]
        } else {
          vT[(bh * HD + d) * NPAD + n_] = (__bf16)val;             // transposed [64, 224]
        }
      }
    }
}

// ---------------- Kernel 2: attention (S = qk^T + bias, softmax, P@V) ----------------
__global__ __launch_bounds__(128) void attn_kernel(
    const __bf16* __restrict__ qs, const __bf16* __restrict__ kn,
    const __bf16* __restrict__ vT, const float* __restrict__ table,
    const int* __restrict__ ridx, __bf16* __restrict__ y) {
  __shared__ __bf16 pbuf[4][16 * NPAD];   // per-wave P staging (28 KB total)

  const int lane = threadIdx.x & 31;
  const int w    = threadIdx.x >> 5;
  const int mt   = blockIdx.x * 4 + w;     // row tile, 0..15
  if (mt >= 13) return;                    // 13 = ceil(197/16); no barriers used
  const int h = blockIdx.y, b = blockIdx.z;
  const int l15 = lane & 15, h4 = lane >> 4;

  const size_t bh = (size_t)(b * NHEAD + h);
  const __bf16* qsb = qs + bh * NTOK * HD;
  const __bf16* knb = kn + bh * NTOK * HD;
  const __bf16* vTb = vT + bh * HD * NPAD;

  // ---- S = q @ k^T over K = 64 (2 WMMA steps), 13 column tiles ----
  v8f s[13];
#pragma unroll
  for (int t = 0; t < 13; ++t) s[t] = zero8();

  const int arow = mt * 16 + l15;
  const bool arv = arow < NTOK;
  const __bf16* qrow = qsb + (size_t)(arv ? arow : 0) * HD;
  v16bf aq0 = a_frag_bf16(qrow, 0,  lane, arv);
  v16bf aq1 = a_frag_bf16(qrow, 32, lane, arv);

#pragma unroll
  for (int t = 0; t < 13; ++t) {
    int ncol = t * 16 + l15;
    bool cv = ncol < NTOK;
    int nc = cv ? ncol : 0;
    // B[k][n] = K[n][k]: natural [tok,64] layout -> contiguous per-lane loads
    v16bf b0 = b_frag_bf16(knb, HD, nc, 0,  lane, cv);
    s[t] = wmma_bf(aq0, b0, s[t]);
    v16bf b1 = b_frag_bf16(knb, HD, nc, 32, lane, cv);
    s[t] = wmma_bf(aq1, b1, s[t]);
  }

  // ---- relative position bias + mask ----
#pragma unroll
  for (int t = 0; t < 13; ++t) {
    int ncol = t * 16 + l15;
#pragma unroll
    for (int r = 0; r < 8; ++r) {
      int nrow = mt * 16 + r + 8 * h4;
      float sv = s[t][r];
      if (nrow < NTOK && ncol < NTOK)
        sv += table[ridx[nrow * NTOK + ncol] * NHEAD + h];
      else
        sv = -1e30f;
      s[t][r] = sv;
    }
  }

  // ---- softmax across the 208-wide row strip (16-lane half-wave reductions) ----
#pragma unroll
  for (int r = 0; r < 8; ++r) {
    float m = s[0][r];
#pragma unroll
    for (int t = 1; t < 13; ++t) m = fmaxf(m, s[t][r]);
    m = fmaxf(m, __shfl_xor(m, 1, 32));
    m = fmaxf(m, __shfl_xor(m, 2, 32));
    m = fmaxf(m, __shfl_xor(m, 4, 32));
    m = fmaxf(m, __shfl_xor(m, 8, 32));
    float sum = 0.f;
#pragma unroll
    for (int t = 0; t < 13; ++t) { float p = __expf(s[t][r] - m); s[t][r] = p; sum += p; }
    sum += __shfl_xor(sum, 1, 32);
    sum += __shfl_xor(sum, 2, 32);
    sum += __shfl_xor(sum, 4, 32);
    sum += __shfl_xor(sum, 8, 32);
    float inv = 1.f / sum;
#pragma unroll
    for (int t = 0; t < 13; ++t) s[t][r] *= inv;
  }

  // ---- stage P (bf16) into per-wave LDS, zero-pad cols 208..223 ----
  __bf16* pb = pbuf[w];
#pragma unroll
  for (int t = 0; t < 13; ++t)
#pragma unroll
    for (int r = 0; r < 8; ++r)
      pb[(r + 8 * h4) * NPAD + t * 16 + l15] = (__bf16)s[t][r];
#pragma unroll
  for (int i = 0; i < 8; ++i) {
    int e = lane * 8 + i;                       // 0..255 -> 16x16 pad block
    pb[(e >> 4) * NPAD + 208 + (e & 15)] = (__bf16)0.f;
  }

  // ---- O = P @ V : K = 224 (7 steps), N = 64 (4 tiles); V^T is zero-padded ----
  v8f o[4];
#pragma unroll
  for (int t = 0; t < 4; ++t) o[t] = zero8();
  const __bf16* prow = pb + (size_t)l15 * NPAD;
  for (int kt = 0; kt < NPAD / 32; ++kt) {
    v16bf ap = a_frag_bf16(prow, kt * 32, lane, true);
#pragma unroll
    for (int nt = 0; nt < 4; ++nt) {
      // B[k][n] = V[k][n] = V^T[n][k]: contiguous per-lane loads, no guards
      v16bf bv = b_frag_bf16(vTb, NPAD, nt * 16 + l15, kt * 32, lane, true);
      o[nt] = wmma_bf(ap, bv, o[nt]);
    }
  }

  // ---- epilogue: y[b, tok, h*64 + d] (bf16) ----
#pragma unroll
  for (int nt = 0; nt < 4; ++nt)
#pragma unroll
    for (int r = 0; r < 8; ++r) {
      int tok = mt * 16 + r + 8 * h4;
      if (tok < NTOK)
        y[(size_t)(b * NTOK + tok) * DIMC + h * HD + nt * 16 + l15] = (__bf16)o[nt][r];
    }
}

// ---------------- Kernel 3: output projection ----------------
// out[12608, 768] = y @ proj_w^T + proj_b  (f32 out); 2 m-tiles x 4 n-tiles per wave.
__global__ __launch_bounds__(128) void proj_kernel(
    const __bf16* __restrict__ y, const __bf16* __restrict__ wproj,
    const float* __restrict__ proj_b, float* __restrict__ out) {
  const int lane = threadIdx.x & 31;
  const int w    = threadIdx.x >> 5;
  const int mt0  = blockIdx.x * 2;             // 0..787 (two tiles)
  const int nt0  = (blockIdx.y * 4 + w) * 4;   // 4 n-tiles per wave (48 total)
  const int l15  = lane & 15, h4 = lane >> 4;

  v8f acc[2][4];
#pragma unroll
  for (int m = 0; m < 2; ++m)
#pragma unroll
    for (int t = 0; t < 4; ++t) acc[m][t] = zero8();

  const __bf16* yrow0 = y + (size_t)(mt0 * 16 + l15) * DIMC;
  const __bf16* yrow1 = yrow0 + (size_t)16 * DIMC;
  for (int kt = 0; kt < DIMC / 32; ++kt) {
    v16bf a0 = a_frag_bf16(yrow0, kt * 32, lane, true);
    v16bf a1 = a_frag_bf16(yrow1, kt * 32, lane, true);
#pragma unroll
    for (int t = 0; t < 4; ++t) {
      int n = (nt0 + t) * 16 + l15;
      v16bf b = b_frag_bf16(wproj, DIMC, n, kt * 32, lane, true);
      acc[0][t] = wmma_bf(a0, b, acc[0][t]);
      acc[1][t] = wmma_bf(a1, b, acc[1][t]);
    }
  }

#pragma unroll
  for (int m = 0; m < 2; ++m)
#pragma unroll
    for (int t = 0; t < 4; ++t) {
      int c = (nt0 + t) * 16 + l15;
      float pb_ = proj_b[c];
#pragma unroll
      for (int r = 0; r < 8; ++r) {
        int tok = (mt0 + m) * 16 + r + 8 * h4;   // < 12608
        out[(size_t)tok * DIMC + c] = acc[m][t][r] + pb_;
      }
    }
}

// ---------------- Host launcher ----------------
extern "C" void kernel_launch(void* const* d_in, const int* in_sizes, int n_in,
                              void* d_out, int out_size, void* d_ws, size_t ws_size,
                              hipStream_t stream) {
  const float* x       = (const float*)d_in[0];
  const float* qkv_w   = (const float*)d_in[1];
  const float* q_bias  = (const float*)d_in[2];
  const float* v_bias  = (const float*)d_in[3];
  const float* table   = (const float*)d_in[4];
  const float* proj_w  = (const float*)d_in[5];
  const float* proj_b  = (const float*)d_in[6];
  const int*   ridx    = (const int*)d_in[7];
  float* out = (float*)d_out;

  char* p = (char*)d_ws;
  __bf16* qs = (__bf16*)p; p += (size_t)NBATCH * NHEAD * NTOK * HD * 2;   // 19.4 MB
  __bf16* kn = (__bf16*)p; p += (size_t)NBATCH * NHEAD * NTOK * HD * 2;   // 19.4 MB
  __bf16* vT = (__bf16*)p; p += (size_t)NBATCH * NHEAD * HD * NPAD * 2;   // 22.0 MB
  __bf16* yb = (__bf16*)p; p += (size_t)MROWS * DIMC * 2;                 // 19.4 MB
  __bf16* xb = (__bf16*)p; p += (size_t)MROWS * DIMC * 2;                 // 19.4 MB
  __bf16* wqkvb  = (__bf16*)p; p += (size_t)3 * DIMC * DIMC * 2;          //  3.5 MB
  __bf16* wprojb = (__bf16*)p; p += (size_t)DIMC * DIMC * 2;              //  1.2 MB

  // C) one-time-per-launch f32 -> bf16 conversions (all counts divisible by 8)
  {
    const int nx = MROWS * DIMC;            // 9,682,944
    const int nq = 3 * DIMC * DIMC;         // 1,769,472
    const int np = DIMC * DIMC;             //   589,824
    cvt_kernel<<<dim3(nx / 2048), 256, 0, stream>>>(x, xb, nx);
    cvt_kernel<<<dim3(nq / 2048), 256, 0, stream>>>(qkv_w, wqkvb, nq);
    cvt_kernel<<<dim3(np / 2048), 256, 0, stream>>>(proj_w, wprojb, np);
  }

  // 0) zero V^T token-padding columns (PV K-loop then needs no bounds checks)
  {
    const int total = NBATCH * NHEAD * HD * (NPAD - NTOK);
    pad_kernel<<<dim3((total + 255) / 256), 256, 0, stream>>>(vT);
  }

  // 1) QKV GEMM: 394 x 9 blocks, each wave 2 m-tiles x 4 n-tiles
  qkv_kernel<<<dim3(MTILES / 2, 9), 128, 0, stream>>>(xb, wqkvb, q_bias, v_bias, qs, kn, vT);

  // 2) Attention: 13 row-tiles per (b,h), 4 waves per block
  attn_kernel<<<dim3(4, NHEAD, NBATCH), 128, 0, stream>>>(qs, kn, vT, table, ridx, yb);

  // 3) Projection: 394 x 3 blocks, each wave 2 m-tiles x 4 n-tiles
  proj_kernel<<<dim3(MTILES / 2, 3), 128, 0, stream>>>(yb, wprojb, proj_b, out);
}
